// HybridProcessor_47476568490198
// MI455X (gfx1250) — matrix operand order repaired and checked by reading, not compiled, not run on hardware
//
#include <hip/hip_runtime.h>

// ---------------------------------------------------------------------------
// Types / feature detection
// ---------------------------------------------------------------------------
typedef __attribute__((ext_vector_type(16))) _Float16 v16h;
typedef __attribute__((ext_vector_type(8)))  _Float16 v8h;
typedef __attribute__((ext_vector_type(8)))  float    v8f;
typedef __attribute__((ext_vector_type(4)))  int      v4i;

#if defined(__has_builtin)
#if __has_builtin(__builtin_amdgcn_global_load_async_to_lds_b128)
#define HAVE_ASYNC_LDS 1
#endif
#endif

typedef __attribute__((address_space(1))) v4i as1_v4i;  // global
typedef __attribute__((address_space(3))) v4i as3_v4i;  // LDS

// copy 16 contiguous bytes global -> LDS (async DMA when available)
__device__ __forceinline__ void async_copy16(const _Float16* g, _Float16* l) {
#if defined(HAVE_ASYNC_LDS)
  __builtin_amdgcn_global_load_async_to_lds_b128((as1_v4i*)g, (as3_v4i*)l, 0, 0);
#else
  *(v8h*)l = *(const v8h*)g;
#endif
}

__device__ __forceinline__ void wait_async() {
#if defined(HAVE_ASYNC_LDS)
#if __has_builtin(__builtin_amdgcn_s_wait_asynccnt)
  __builtin_amdgcn_s_wait_asynccnt(0);
#else
  asm volatile("s_wait_asynccnt 0x0" ::: "memory");
#endif
#endif
}

#define BM 128
#define BN 256
#define BK 32
#define LDT 40  // halves per LDS tile row (32 + 8 pad) -> 80B stride, 16B aligned

// ---------------------------------------------------------------------------
// GEMM: out[M,N] = epi(A[M,K](f16) @ Wt[N,K](f16)^T + bias[N]) (+resid)
// Wt is the weight pre-transposed to N-major so K is contiguous for both tiles.
// ACT: 0 none, 1 exact gelu, 2 sigmoid.  STOREF: write f32 out. STOREH: f16 out.
// ---------------------------------------------------------------------------
template <int ACT, bool RESID, bool STOREF, bool STOREH>
__global__ __launch_bounds__(256) void gemm_wmma_f16(
    const _Float16* __restrict__ A, const _Float16* __restrict__ Wt,
    const float* __restrict__ bias, float* __restrict__ outF,
    _Float16* __restrict__ outH, const float* __restrict__ resid,
    int M, int N, int K)
{
  __shared__ _Float16 As[BM * LDT];  // 10 KB
  __shared__ _Float16 Bs[BN * LDT];  // 20 KB

  const int tid  = threadIdx.x;
  const int lane = tid & 31;
  const int wave = tid >> 5;
  const int wm   = wave & 1;   // 2 row-groups of 64
  const int wn   = wave >> 1;  // 4 col-groups of 64
  const int blockRow = blockIdx.y * BM;
  const int blockCol = blockIdx.x * BN;

  const v8f zacc = {0.f, 0.f, 0.f, 0.f, 0.f, 0.f, 0.f, 0.f};
  v8f acc[4][4];
#pragma unroll
  for (int i = 0; i < 4; ++i)
#pragma unroll
    for (int j = 0; j < 4; ++j) acc[i][j] = zacc;

  union V16 { v16h v; v8h h[2]; };

  for (int k0 = 0; k0 < K; k0 += BK) {
    // ---- stage A: 128 rows x 32 halves = 512 x 16B chunks, 2 per thread ----
#pragma unroll
    for (int i = 0; i < 2; ++i) {
      int s = tid + i * 256;
      int r = s >> 2;
      int c = (s & 3) << 3;
      async_copy16(A + (size_t)(blockRow + r) * K + (k0 + c), &As[r * LDT + c]);
    }
    // ---- stage B: 256 rows x 32 halves = 1024 x 16B chunks, 4 per thread ----
#pragma unroll
    for (int i = 0; i < 4; ++i) {
      int s = tid + i * 256;
      int r = s >> 2;
      int c = (s & 3) << 3;
      async_copy16(Wt + (size_t)(blockCol + r) * K + (k0 + c), &Bs[r * LDT + c]);
    }
    wait_async();
    __syncthreads();

    // ---- fragments (CDNA5 ISA 7.12.2 wave32 16-bit layouts) ----
    V16 afrag[4];
#pragma unroll
    for (int mt = 0; mt < 4; ++mt) {
      int m  = wm * 64 + mt * 16 + (lane & 15);
      int hb = (lane >> 4) << 3;  // K base 0 / 8
      afrag[mt].h[0] = *(const v8h*)&As[m * LDT + hb];       // K = hb..hb+7
      afrag[mt].h[1] = *(const v8h*)&As[m * LDT + hb + 16];  // K = hb+16..+23
    }
    V16 bfrag[4];
#pragma unroll
    for (int nt = 0; nt < 4; ++nt) {
      int n  = wn * 64 + nt * 16 + (lane & 15);
      int kb = (lane >> 4) << 3;  // K base 0 / 8
      bfrag[nt].h[0] = *(const v8h*)&Bs[n * LDT + kb];
      bfrag[nt].h[1] = *(const v8h*)&Bs[n * LDT + kb + 16];
    }
#pragma unroll
    for (int mt = 0; mt < 4; ++mt)
#pragma unroll
      for (int nt = 0; nt < 4; ++nt)
        acc[mt][nt] = __builtin_amdgcn_wmma_f32_16x16x32_f16(
            false, afrag[mt].v, false, bfrag[nt].v, (short)0, acc[mt][nt],
            false, false);
    __syncthreads();
  }

  // ---- epilogue: D layout: VGPR r -> m = r + 8*(lane>=16), n = lane&15 ----
  const int rOff = (lane >> 4) * 8;
  const int cOff = lane & 15;
#pragma unroll
  for (int mt = 0; mt < 4; ++mt) {
#pragma unroll
    for (int nt = 0; nt < 4; ++nt) {
      int col = blockCol + wn * 64 + nt * 16 + cOff;
      float bv = bias[col];
#pragma unroll
      for (int r = 0; r < 8; ++r) {
        int row = blockRow + wm * 64 + mt * 16 + rOff + r;
        float v = acc[mt][nt][r] + bv;
        if (ACT == 1) v = 0.5f * v * (1.0f + erff(v * 0.70710678118654752f));
        if (ACT == 2) v = 1.0f / (1.0f + expf(-v));
        if (RESID) v += resid[(size_t)row * N + col];
        if (STOREF) outF[(size_t)row * N + col] = v;
        if (STOREH) outH[(size_t)row * N + col] = (_Float16)v;
      }
    }
  }
}

// ---------------------------------------------------------------------------
// LayerNorm over H=1024. outF = LN(in)*g + b (+res post-LN); optional f16 copy.
// ---------------------------------------------------------------------------
template <bool ADD_RES, bool STOREH>
__global__ __launch_bounds__(256) void ln_kernel(
    const float* __restrict__ in, const float* __restrict__ g,
    const float* __restrict__ beta, const float* __restrict__ res,
    float* __restrict__ outF, _Float16* __restrict__ outH)
{
  __shared__ float red[256];
  const int row = blockIdx.x;
  const float* rp = in + (size_t)row * 1024;
  float vals[4];
  float s = 0.f;
#pragma unroll
  for (int i = 0; i < 4; ++i) {
    vals[i] = rp[threadIdx.x + i * 256];
    s += vals[i];
  }
  red[threadIdx.x] = s;
  __syncthreads();
  for (int off = 128; off > 0; off >>= 1) {
    if (threadIdx.x < off) red[threadIdx.x] += red[threadIdx.x + off];
    __syncthreads();
  }
  const float mu = red[0] * (1.0f / 1024.0f);
  __syncthreads();
  float s2 = 0.f;
#pragma unroll
  for (int i = 0; i < 4; ++i) {
    float d = vals[i] - mu;
    s2 += d * d;
  }
  red[threadIdx.x] = s2;
  __syncthreads();
  for (int off = 128; off > 0; off >>= 1) {
    if (threadIdx.x < off) red[threadIdx.x] += red[threadIdx.x + off];
    __syncthreads();
  }
  const float rstd = rsqrtf(red[0] * (1.0f / 1024.0f) + 1e-5f);
#pragma unroll
  for (int i = 0; i < 4; ++i) {
    int c = threadIdx.x + i * 256;
    float v = (vals[i] - mu) * rstd * g[c] + beta[c];
    if (ADD_RES) v += res[(size_t)row * 1024 + c];
    outF[(size_t)row * 1024 + c] = v;
    if (STOREH) outH[(size_t)row * 1024 + c] = (_Float16)v;
  }
}

// ---------------------------------------------------------------------------
// Gated EMA scan per (b, channel): s_t = d*s+(1-d)*h ; outh = f16(Cs*s + D*h)
// ---------------------------------------------------------------------------
__global__ __launch_bounds__(256) void scan_ssm(
    const float* __restrict__ h, const float* __restrict__ dt,
    const float* __restrict__ csum, const float* __restrict__ D,
    _Float16* __restrict__ outh)
{
  const int idx = blockIdx.x * blockDim.x + threadIdx.x;  // 0..B*H-1
  const int b = idx >> 10;
  const int c = idx & 1023;
  const float cs = csum[c], dd = D[c];
  const size_t base = (size_t)b * 2048 * 1024 + c;
  float s = h[base];
  outh[base] = (_Float16)(cs * s + dd * s);
  for (int t = 1; t < 2048; ++t) {
    const size_t p = base + (size_t)t * 1024;
    const float hv = h[p];
    const float d  = dt[p];
    s = d * s + (1.0f - d) * hv;
    outh[p] = (_Float16)(cs * s + dd * hv);
  }
}

// ---------------------------------------------------------------------------
// Linear-attention cumulative scan; outh = f16((kvcum/(kcum+eps))*q)
// ---------------------------------------------------------------------------
__global__ __launch_bounds__(256) void attn_scan(
    const float* __restrict__ q, const float* __restrict__ k,
    const float* __restrict__ v, _Float16* __restrict__ outh)
{
  const int idx = blockIdx.x * blockDim.x + threadIdx.x;
  const int b = idx >> 10;
  const int c = idx & 1023;
  const size_t base = (size_t)b * 2048 * 1024 + c;
  float kc = 0.f, kvc = 0.f;
  for (int t = 0; t < 2048; ++t) {
    const size_t p = base + (size_t)t * 1024;
    const float kv = k[p];
    kc += kv;
    kvc += kv * v[p];
    outh[p] = (_Float16)((kvc / (kc + 1e-8f)) * q[p]);
  }
}

// ---------------------------------------------------------------------------
// Small helpers
// ---------------------------------------------------------------------------
__global__ __launch_bounds__(256) void combine_kernel(
    const float* __restrict__ a, const float* __restrict__ b,
    _Float16* __restrict__ o)
{
  const size_t i = (size_t)blockIdx.x * blockDim.x + threadIdx.x;
  o[i] = (_Float16)(0.5f * (a[i] + b[i]));
}

__global__ __launch_bounds__(256) void f32_to_f16_kernel(
    const float* __restrict__ s, _Float16* __restrict__ d, int n)
{
  const int i = blockIdx.x * blockDim.x + threadIdx.x;
  if (i < n) d[i] = (_Float16)s[i];
}

// transpose-convert: src[K,N] f32 -> dst[N,K] f16 (32x32 tiles via LDS)
__global__ __launch_bounds__(256) void transpose_cvt_kernel(
    const float* __restrict__ src, _Float16* __restrict__ dst, int K, int N)
{
  __shared__ _Float16 tile[32][33];
  const int tx = threadIdx.x;  // 0..31
  const int ty = threadIdx.y;  // 0..7
  const int kb = blockIdx.y * 32;
  const int nb = blockIdx.x * 32;
#pragma unroll
  for (int i = 0; i < 4; ++i)
    tile[ty + 8 * i][tx] =
        (_Float16)src[(size_t)(kb + ty + 8 * i) * N + (nb + tx)];
  __syncthreads();
#pragma unroll
  for (int i = 0; i < 4; ++i)
    dst[(size_t)(nb + ty + 8 * i) * K + (kb + tx)] = tile[tx][ty + 8 * i];
}

__global__ __launch_bounds__(256) void rowsum_kernel(
    const float* __restrict__ C, float* __restrict__ out)
{
  const int h = blockIdx.x * blockDim.x + threadIdx.x;  // 0..1023
  float s = 0.f;
  for (int j = 0; j < 512; ++j) s += C[(size_t)h * 512 + j];
  out[h] = s;
}

// ---------------------------------------------------------------------------
// Launcher
// ---------------------------------------------------------------------------
extern "C" void kernel_launch(void* const* d_in, const int* in_sizes, int n_in,
                              void* d_out, int out_size, void* d_ws,
                              size_t ws_size, hipStream_t stream)
{
  (void)in_sizes; (void)n_in; (void)out_size; (void)ws_size;
  const int H = 1024, H2 = 2048, Bsz = 8, L = 2048;
  const int M = Bsz * L;            // 16384
  const size_t MH = (size_t)M * H;  // 16.7M
  const size_t HH = (size_t)H * H;

  const float* x      = (const float*)d_in[0];
  const float* W_in   = (const float*)d_in[1];
  const float* b_in   = (const float*)d_in[2];
  const float* lW1    = (const float*)d_in[3];
  const float* lb1    = (const float*)d_in[4];
  const float* lW2    = (const float*)d_in[5];
  const float* lb2    = (const float*)d_in[6];
  const float* lg     = (const float*)d_in[7];
  const float* lbt    = (const float*)d_in[8];
  const float* Wdt    = (const float*)d_in[9];
  const float* bdt    = (const float*)d_in[10];
  const float* Cmat   = (const float*)d_in[11];
  const float* Dvec   = (const float*)d_in[12];
  const float* Wo_ssm = (const float*)d_in[13];
  const float* bo_ssm = (const float*)d_in[14];
  const float* Wq     = (const float*)d_in[15];
  const float* bq     = (const float*)d_in[16];
  const float* Wk     = (const float*)d_in[17];
  const float* bk     = (const float*)d_in[18];
  const float* Wv     = (const float*)d_in[19];
  const float* bv     = (const float*)d_in[20];
  const float* Wo_att = (const float*)d_in[21];
  const float* bo_att = (const float*)d_in[22];
  const float* g_att  = (const float*)d_in[23];
  const float* be_att = (const float*)d_in[24];
  const float* Wout   = (const float*)d_in[25];
  const float* bout   = (const float*)d_in[26];
  const float* g_out  = (const float*)d_in[27];
  const float* be_out = (const float*)d_in[28];

  // workspace: 5 f32 activation slots + f16 mirrors + transposed f16 weights
  float* fws = (float*)d_ws;
  float* s1 = fws;            // [M,H]  h / ssm_out
  float* s2 = s1 + MH;        // [M,H]  y / dt / final pre-LN
  float* s3 = s2 + MH;        // [M,H]  q
  float* s4 = s3 + MH;        // [M,H]  k / att pre-LN
  float* s5 = s4 + MH;        // [M,H]  v / att_out
  float* csumb = s5 + MH;     // [H]
  _Float16* xh  = (_Float16*)(csumb + H);  // f16 mirror of x
  _Float16* hh  = xh + MH;                 // f16 mirror of h (also 'combined')
  _Float16* t2h = hh + MH;                 // [M,2H] f16 (gelu out / ssm_in / attq)
  _Float16* wts = t2h + 2 * MH;            // transposed weights, N-major

  _Float16* wt_in   = wts;
  _Float16* wt_l1   = wt_in + HH;
  _Float16* wt_l2   = wt_l1 + 4 * (size_t)H * H2;
  _Float16* wt_dt   = wt_l2 + 4 * (size_t)H2 * H;
  _Float16* wt_oss  = wt_dt + HH;
  _Float16* wt_q    = wt_oss + HH;
  _Float16* wt_k    = wt_q + HH;
  _Float16* wt_v    = wt_k + HH;
  _Float16* wt_oatt = wt_v + HH;
  _Float16* wt_out  = wt_oatt + HH;

  // --- one-time conversions ---
  f32_to_f16_kernel<<<dim3((unsigned)((MH + 255) / 256)), 256, 0, stream>>>(
      x, xh, (int)MH);
  auto tcvt = [&](const float* s, _Float16* d, int K, int N) {
    transpose_cvt_kernel<<<dim3(N / 32, K / 32), dim3(32, 8), 0, stream>>>(
        s, d, K, N);
  };
  tcvt(W_in, wt_in, H, H);
  for (int i = 0; i < 4; ++i)
    tcvt(lW1 + (size_t)i * H * H2, wt_l1 + (size_t)i * H * H2, H, H2);
  for (int i = 0; i < 4; ++i)
    tcvt(lW2 + (size_t)i * H2 * H, wt_l2 + (size_t)i * H2 * H, H2, H);
  tcvt(Wdt, wt_dt, H, H);
  tcvt(Wo_ssm, wt_oss, H, H);
  tcvt(Wq, wt_q, H, H);
  tcvt(Wk, wt_k, H, H);
  tcvt(Wv, wt_v, H, H);
  tcvt(Wo_att, wt_oatt, H, H);
  tcvt(Wout, wt_out, H, H);
  rowsum_kernel<<<dim3(4), 256, 0, stream>>>(Cmat, csumb);

  const dim3 blk(256);
  const dim3 gN1(H / BN, M / BM);   // N=1024 -> (4,128)
  const dim3 gN2(H2 / BN, M / BM);  // N=2048 -> (8,128)
  const dim3 gScan((Bsz * H) / 256);

  // ---- SSM branch ----
  gemm_wmma_f16<0, false, true, true><<<gN1, blk, 0, stream>>>(
      xh, wt_in, b_in, s1, hh, nullptr, M, H, H);
  for (int i = 0; i < 4; ++i) {
    gemm_wmma_f16<1, false, false, true><<<gN2, blk, 0, stream>>>(
        hh, wt_l1 + (size_t)i * H * H2, lb1 + (size_t)i * H2, nullptr, t2h,
        nullptr, M, H2, H);
    gemm_wmma_f16<0, false, true, false><<<gN1, blk, 0, stream>>>(
        t2h, wt_l2 + (size_t)i * H2 * H, lb2 + (size_t)i * H, s2, nullptr,
        nullptr, M, H, H2);
    ln_kernel<true, true><<<dim3(M), blk, 0, stream>>>(
        s2, lg + (size_t)i * H, lbt + (size_t)i * H, s1, s1, hh);
  }
  gemm_wmma_f16<2, false, true, false><<<gN1, blk, 0, stream>>>(
      hh, wt_dt, bdt, s2, nullptr, nullptr, M, H, H);       // dt -> s2
  scan_ssm<<<gScan, blk, 0, stream>>>(s1, s2, csumb, Dvec, t2h);  // ssm_in f16
  gemm_wmma_f16<0, false, true, false><<<gN1, blk, 0, stream>>>(
      t2h, wt_oss, bo_ssm, s1, nullptr, nullptr, M, H, H);  // ssm_out -> s1

  // ---- linear attention branch ----
  gemm_wmma_f16<0, false, true, false><<<gN1, blk, 0, stream>>>(
      xh, wt_q, bq, s3, nullptr, nullptr, M, H, H);
  gemm_wmma_f16<0, false, true, false><<<gN1, blk, 0, stream>>>(
      xh, wt_k, bk, s4, nullptr, nullptr, M, H, H);
  gemm_wmma_f16<0, false, true, false><<<gN1, blk, 0, stream>>>(
      xh, wt_v, bv, s5, nullptr, nullptr, M, H, H);
  attn_scan<<<gScan, blk, 0, stream>>>(s3, s4, s5, t2h + MH);  // attq f16
  gemm_wmma_f16<0, true, true, false><<<gN1, blk, 0, stream>>>(
      t2h + MH, wt_oatt, bo_att, s4, nullptr, x, M, H, H);  // + x resid
  ln_kernel<false, false><<<dim3(M), blk, 0, stream>>>(
      s4, g_att, be_att, nullptr, s5, nullptr);             // att_out -> s5

  // ---- combine + output ----
  combine_kernel<<<dim3((unsigned)(MH / 256)), blk, 0, stream>>>(s1, s5, hh);
  gemm_wmma_f16<0, false, true, false><<<gN1, blk, 0, stream>>>(
      hh, wt_out, bout, s2, nullptr, nullptr, M, H, H);
  ln_kernel<false, false><<<dim3(M), blk, 0, stream>>>(
      s2, g_out, be_out, nullptr, (float*)d_out, nullptr);
}
